// SNNNetwork_8650064134923
// MI455X (gfx1250) — matrix operand order repaired and checked by reading, not compiled, not run on hardware
//
#include <hip/hip_runtime.h>
#include <hip/hip_bf16.h>

typedef __attribute__((ext_vector_type(16))) _Float16 v16h;
typedef __attribute__((ext_vector_type(8)))  _Float16 v8h;
typedef __attribute__((ext_vector_type(8)))  float    v8f;

// ---------------------------------------------------------------------------
// Stage 1: convert fp32 weights -> fp16 packed into workspace.
// Layout (in halfs): W1 @0 (256*128), W2 @32768 (128*256), W3 @65536 (64*128),
//                    W4 @73728 (64*64). Total 77824 halfs = 152 KB.
// ---------------------------------------------------------------------------
__global__ void snn_convert_weights(const float* __restrict__ W1,
                                    const float* __restrict__ W2,
                                    const float* __restrict__ W3,
                                    const float* __restrict__ W4,
                                    _Float16* __restrict__ dst) {
    int i = blockIdx.x * blockDim.x + threadIdx.x;
    if (i < 32768)          dst[i] = (_Float16)W1[i];
    else if (i < 65536)     dst[i] = (_Float16)W2[i - 32768];
    else if (i < 73728)     dst[i] = (_Float16)W3[i - 65536];
    else if (i < 77824)     dst[i] = (_Float16)W4[i - 73728];
}

// ---------------------------------------------------------------------------
// Helpers
// ---------------------------------------------------------------------------
static __device__ __forceinline__ unsigned snn_hash3(unsigned a, unsigned b, unsigned c) {
    unsigned h = a * 0x9E3779B1u ^ b * 0x85EBCA77u ^ c * 0xC2B2AE3Du;
    h ^= h >> 16; h *= 0x7FEB352Du;
    h ^= h >> 15; h *= 0x846CA68Bu;
    h ^= h >> 16;
    return h;
}

// 16-bit A fragment (16x32, MxK) from LDS, row-major [16][stride] halfs.
// lanes 0-15 : M=lane,    K = kb+{0..7}  and kb+{16..23}
// lanes 16-31: M=lane-16, K = kb+{8..15} and kb+{24..31}
static __device__ __forceinline__ v16h snn_load_A(const _Float16* buf, int stride,
                                                  int kb, int lane) {
    const int m    = lane & 15;
    const int koff = (lane >> 4) * 8;
    const v8h lo = *(const v8h*)(buf + m * stride + kb + koff);
    const v8h hi = *(const v8h*)(buf + m * stride + kb + 16 + koff);
    v16h a;
#pragma unroll
    for (int i = 0; i < 8; ++i) { a[i] = lo[i]; a[8 + i] = hi[i]; }
    return a;
}

// 16-bit B fragment (32x16, KxN) from global f16 weights, torch layout (O, I):
// B[k][n] = W[nbase+n][kb+k].
// lanes 0-15 : N=lane,    K = kb+{0..7} and kb+{16..23} ; lanes 16-31: N=lane-16, K = kb+{8..15} and kb+{24..31}
static __device__ __forceinline__ v16h snn_load_B(const _Float16* __restrict__ W, int I,
                                                  int nbase, int kb, int lane) {
    const int n = nbase + (lane & 15);
    const int k = kb + (lane >> 4) * 8;
    const v8h* p = (const v8h*)(W + n * I + k);
    const v8h lo = p[0];
    const v8h hi = p[2];
    v16h b;
#pragma unroll
    for (int i = 0; i < 8; ++i) { b[i] = lo[i]; b[8 + i] = hi[i]; }
    return b;
}

// C = spikes(16 x K, LDS) * W_tile^T (+ bias broadcast over rows)
static __device__ __forceinline__ v8f snn_tile_matmul(const _Float16* sbuf, int stride, int K,
                                                      const _Float16* __restrict__ W, int I,
                                                      int nbase, const float* __restrict__ bias,
                                                      int lane) {
    const float bv = bias[nbase + (lane & 15)];
    v8f acc;
#pragma unroll
    for (int r = 0; r < 8; ++r) acc[r] = bv;
#pragma unroll
    for (int kb = 0; kb < K; kb += 32) {
        v16h a = snn_load_A(sbuf, stride, kb, lane);
        v16h b = snn_load_B(W, I, nbase, kb, lane);
        acc = __builtin_amdgcn_wmma_f32_16x16x32_f16(false, a, false, b,
                                                     (short)0, acc, false, false);
    }
    return acc;
}

// LIF update (tau=2, v_th=1, hard reset to 0) on the 16x16 C tile; spike -> LDS f16.
// C/D layout: VGPR r holds (M = r + 8*(lane>=16), N = lane%16).
static __device__ __forceinline__ void snn_lif_store(v8f& v, v8f c, _Float16* sout,
                                                     int stride, int nbase, int lane) {
    const int n     = nbase + (lane & 15);
    const int mbase = (lane >> 4) * 8;
#pragma unroll
    for (int r = 0; r < 8; ++r) {
        float vv = v[r] + (c[r] - v[r]) * 0.5f;
        float s  = (vv >= 1.0f) ? 1.0f : 0.0f;
        v[r]     = vv * (1.0f - s);
        sout[(mbase + r) * stride + n] = (_Float16)s;
    }
}

// ---------------------------------------------------------------------------
// Fully-fused SNN kernel. One block = 256 threads = 8 wave32 = 16 batch rows.
// t is the outer loop; LIF membrane states live in WMMA accumulator VGPRs.
// ---------------------------------------------------------------------------
__global__ __launch_bounds__(256) void snn_fused(const float* __restrict__ x,
                                                 const _Float16* __restrict__ wf16,
                                                 const float* __restrict__ b1,
                                                 const float* __restrict__ b2,
                                                 const float* __restrict__ b3,
                                                 const float* __restrict__ b4,
                                                 float* __restrict__ out) {
    __shared__ _Float16 sIn[16 * 128];  // encoder spikes        (4 KB)
    __shared__ _Float16 s1 [16 * 256];  // layer-1 spikes        (8 KB)
    __shared__ _Float16 s2 [16 * 128];  // layer-2 spikes        (4 KB)
    __shared__ _Float16 s3 [16 * 64];   // layer-3 spikes        (2 KB)

    const int tid     = threadIdx.x;
    const int lane    = tid & 31;
    const int w       = tid >> 5;
    const int rowbase = blockIdx.x * 16;

    const _Float16* W1 = wf16;
    const _Float16* W2 = wf16 + 32768;
    const _Float16* W3 = wf16 + 65536;
    const _Float16* W4 = wf16 + 73728;

    // Preload encoder firing probabilities: 16*128 = 2048 elems, 8 per thread.
    float p[8];
    int   em[8], ei[8];
#pragma unroll
    for (int j = 0; j < 8; ++j) {
        const int e = tid + j * 256;
        em[j] = e >> 7;
        ei[j] = e & 127;
        const float xv = x[(rowbase + em[j]) * 128 + ei[j]];
        p[j] = 1.0f / (1.0f + __expf(-xv));
    }

    // LIF membrane states + output accumulator, resident in VGPRs.
    v8f v1a = {}, v1b = {}, v2 = {}, v3 = {}, acc4 = {};

    for (int t = 0; t < 8; ++t) {
        // -------- encoder: bernoulli(sigmoid(x)) per timestep --------
#pragma unroll
        for (int j = 0; j < 8; ++j) {
            const unsigned h = snn_hash3(42u + (unsigned)t,
                                         (unsigned)(rowbase + em[j]),
                                         (unsigned)ei[j]);
            const float u = (float)(h >> 8) * (1.0f / 16777216.0f);
            sIn[em[j] * 128 + ei[j]] = (_Float16)((u < p[j]) ? 1.0f : 0.0f);
        }
        __syncthreads();

        // -------- layer 1: 128 -> 256, wave w owns tiles w and w+8 --------
        {
            v8f c = snn_tile_matmul(sIn, 128, 128, W1, 128, w * 16, b1, lane);
            snn_lif_store(v1a, c, s1, 256, w * 16, lane);
            c = snn_tile_matmul(sIn, 128, 128, W1, 128, (w + 8) * 16, b1, lane);
            snn_lif_store(v1b, c, s1, 256, (w + 8) * 16, lane);
        }
        __syncthreads();

        // -------- layer 2: 256 -> 128, wave w owns tile w --------
        {
            v8f c = snn_tile_matmul(s1, 256, 256, W2, 256, w * 16, b2, lane);
            snn_lif_store(v2, c, s2, 128, w * 16, lane);
        }
        __syncthreads();

        // -------- layer 3: 128 -> 64, waves 0..3 --------
        if (w < 4) {
            v8f c = snn_tile_matmul(s2, 128, 128, W3, 128, w * 16, b3, lane);
            snn_lif_store(v3, c, s3, 64, w * 16, lane);
        }
        __syncthreads();

        // -------- layer 4: 64 -> 64, accumulate over t --------
        if (w < 4) {
#pragma unroll
            for (int kb = 0; kb < 64; kb += 32) {
                v16h a = snn_load_A(s3, 64, kb, lane);
                v16h b = snn_load_B(W4, 64, w * 16, kb, lane);
                acc4 = __builtin_amdgcn_wmma_f32_16x16x32_f16(false, a, false, b,
                                                              (short)0, acc4, false, false);
            }
        }
        __syncthreads();
    }

    // out = mean_t(h*W4) + b4 = acc/8 + b4
    if (w < 4) {
        const int n     = w * 16 + (lane & 15);
        const int mbase = (lane >> 4) * 8;
        const float bv  = b4[n];
#pragma unroll
        for (int r = 0; r < 8; ++r)
            out[(rowbase + mbase + r) * 64 + n] = acc4[r] * 0.125f + bv;
    }
}

// ---------------------------------------------------------------------------
// Launch: inputs dict order = x, W1, b1, W2, b2, W3, b3, W4, b4
// ---------------------------------------------------------------------------
extern "C" void kernel_launch(void* const* d_in, const int* in_sizes, int n_in,
                              void* d_out, int out_size, void* d_ws, size_t ws_size,
                              hipStream_t stream) {
    (void)in_sizes; (void)n_in; (void)out_size; (void)ws_size;

    const float* x  = (const float*)d_in[0];
    const float* W1 = (const float*)d_in[1];
    const float* b1 = (const float*)d_in[2];
    const float* W2 = (const float*)d_in[3];
    const float* b2 = (const float*)d_in[4];
    const float* W3 = (const float*)d_in[5];
    const float* b3 = (const float*)d_in[6];
    const float* W4 = (const float*)d_in[7];
    const float* b4 = (const float*)d_in[8];

    _Float16* wf16 = (_Float16*)d_ws;
    float*    out  = (float*)d_out;

    snn_convert_weights<<<(77824 + 255) / 256, 256, 0, stream>>>(W1, W2, W3, W4, wf16);

    const int B = 32768;
    snn_fused<<<B / 16, 256, 0, stream>>>(x, wf16, b1, b2, b3, b4, out);
}
